// SF_GCN_Layers_69054484185855
// MI455X (gfx1250) — matrix operand order repaired and checked by reading, not compiled, not run on hardware
//
#include <hip/hip_runtime.h>

#define N_NODES_C 50000
#define N_EDGES_C 800000
#define NUM_GRAPHS_C 64
#define IN_F_C 128
#define FDIM 256  // HID_F == OUT_F == 256

typedef __attribute__((ext_vector_type(16))) __bf16 v16bf;
typedef __attribute__((ext_vector_type(8)))  float  v8f;

__device__ __forceinline__ unsigned bfbits(float f) {
  unsigned u = __float_as_uint(f);
  u += 0x7FFFu + ((u >> 16) & 1u);  // round-to-nearest-even
  return u >> 16;
}
__device__ __forceinline__ unsigned pack2bf(float lo, float hi) {
  return (bfbits(hi) << 16) | bfbits(lo);
}

// Monotone encoding so unsigned atomic-max == float max
__device__ __forceinline__ unsigned fenc(float f) {
  unsigned u = __float_as_uint(f);
  return (u & 0x80000000u) ? ~u : (u | 0x80000000u);
}
__device__ __forceinline__ float fdec(unsigned e) {
  unsigned u = (e & 0x80000000u) ? (e & 0x7FFFFFFFu) : ~e;
  return __uint_as_float(u);
}

__device__ __forceinline__ void atomic_add_f32(float* p, float v) {
  __hip_atomic_fetch_add(p, v, __ATOMIC_RELAXED, __HIP_MEMORY_SCOPE_AGENT);
}

// ---------------------------------------------------------------------------
// H[nrows x 256] = A[nrows x K](f32) @ W[K x 256](f32)  via bf16 WMMA, f32 acc
// Block: 256 threads = 8 waves; block tile 32x128; each wave owns a 16x32
// output slice (2 accumulators sharing one A fragment -> 2 WMMAs / K-step).
// Tiles are staged into LDS in WMMA *fragment order* so each lane loads each
// 16xbf16 operand as one 32B contiguous LDS read (2x ds_load_b128).
// ---------------------------------------------------------------------------
__global__ __launch_bounds__(256) void gemm_wmma_bf16(
    const float* __restrict__ A, const float* __restrict__ W,
    float* __restrict__ H, int nrows, int K) {
  // [tile][lane][8 dwords] == per-lane VGPR image of the A/B fragments
  __shared__ __align__(32) unsigned AF[2][32][8];  // 2 KB  (32 rows x 32 k)
  __shared__ __align__(32) unsigned BF[8][32][8];  // 8 KB  (32 k x 128 n)

  const int tid  = threadIdx.x;
  const int lane = tid & 31;
  const int wid  = tid >> 5;      // 0..7
  const int wrow = wid >> 2;      // 0..1  (16-row tile within block)
  const int wcol = wid & 3;       // 0..3  (32-col slice within block)
  const int half = lane >> 4;     // 0..1
  const int mr   = lane & 15;     // 0..15
  const int row0 = blockIdx.x * 32;
  const int col0 = blockIdx.y * 128;
  const bool full = (row0 + 32 <= nrows);

  v8f acc0 = {};
  v8f acc1 = {};
  for (int k0 = 0; k0 < K; k0 += 32) {
    // ---- stage A tile (32 rows x 32 k) in fragment order: 2 k-pairs/thread
#pragma unroll
    for (int i = 0; i < 2; ++i) {
      int p  = tid + i * 256;       // pair index 0..511
      int ar = p >> 4;              // row in tile 0..31
      int k  = (p & 15) * 2;        // even k 0..30
      float2 av;
      if (full) {
        av = *(const float2*)(A + (size_t)(row0 + ar) * K + (k0 + k));
      } else {
        int gr = row0 + ar;
        av = (gr < nrows) ? *(const float2*)(A + (size_t)gr * K + (k0 + k))
                          : make_float2(0.0f, 0.0f);
      }
      // inverse of ISA 7.12.2 16-bit A 16x32 layout
      int tile = ar >> 4, m = ar & 15;
      int h  = (k >> 3) & 1;        // lane-half holding this k
      int kk = k - 8 * h;
      int j  = (kk >= 16) ? (kk - 8) : kk;  // element slot (even)
      AF[tile][m + 16 * h][j >> 1] = pack2bf(av.x, av.y);
    }
    // ---- stage B tile (32 k x 128 n) in fragment order: 8 k-pairs/thread
#pragma unroll
    for (int i = 0; i < 8; ++i) {
      int p = tid + i * 256;        // pair index 0..2047
      int n = p & 127;              // column in tile (coalesced over threads)
      int k = (p >> 7) * 2;         // even k 0..30
      float v0 = W[(size_t)(k0 + k) * FDIM + (col0 + n)];
      float v1 = W[(size_t)(k0 + k + 1) * FDIM + (col0 + n)];
      // inverse of dense 16-bit B 32x16 layout (table k order, same as A)
      int h = (k >> 3) & 1;
      BF[n >> 4][(n & 15) + 16 * h][((k & 7) + 8 * (k >> 4)) >> 1] = pack2bf(v0, v1);
    }
    __syncthreads();

    const v16bf afrag  = *reinterpret_cast<const v16bf*>(&AF[wrow][lane][0]);
    const v16bf bfrag0 = *reinterpret_cast<const v16bf*>(&BF[2 * wcol][lane][0]);
    const v16bf bfrag1 = *reinterpret_cast<const v16bf*>(&BF[2 * wcol + 1][lane][0]);
    acc0 = __builtin_amdgcn_wmma_f32_16x16x32_bf16(
        false, afrag, false, bfrag0, (short)0, acc0, false, false);
    acc1 = __builtin_amdgcn_wmma_f32_16x16x32_bf16(
        false, afrag, false, bfrag1, (short)0, acc1, false, false);
    __syncthreads();
  }

  // C/D layout: vgpr v, lanes 0-15 -> M=v, lanes 16-31 -> M=v+8; N=lane%16
  const int colbase = col0 + wcol * 32;
#pragma unroll
  for (int v = 0; v < 8; ++v) {
    int gr = row0 + wrow * 16 + v + 8 * half;
    if (gr < nrows) {
      H[(size_t)gr * FDIM + (colbase + mr)]      = acc0[v];
      H[(size_t)gr * FDIM + (colbase + 16 + mr)] = acc1[v];
    }
  }
}

// ---------------------------------------------------------------------------
// Degree / normalization
// ---------------------------------------------------------------------------
__global__ void deg_kernel(const long long* __restrict__ dst,
                           int* __restrict__ degi, int E) {
  int e = blockIdx.x * blockDim.x + threadIdx.x;
  if (e < E)
    __hip_atomic_fetch_add(&degi[(int)dst[e]], 1, __ATOMIC_RELAXED,
                           __HIP_MEMORY_SCOPE_AGENT);
}

__global__ void dinv_kernel(const int* __restrict__ degi,
                            float* __restrict__ dinv, int n) {
  int i = blockIdx.x * blockDim.x + threadIdx.x;
  if (i < n) dinv[i] = rsqrtf((float)degi[i] + 1.0f);
}

// ---------------------------------------------------------------------------
// Edge scatter: agg[dst] += h[src] * dinv[src]*dinv[dst]
// one thread = one (edge, 4-feature chunk); float4 gather + 4 hw f32 atomics
// ---------------------------------------------------------------------------
__global__ __launch_bounds__(256) void edge_agg_kernel(
    const float* __restrict__ h, const long long* __restrict__ src,
    const long long* __restrict__ dst, const float* __restrict__ dinv,
    float* __restrict__ agg, int E) {
  int gid = blockIdx.x * 256 + threadIdx.x;
  int e = gid >> 6;          // edge
  int c = gid & 63;          // feature chunk (4 floats)
  if (e >= E) return;
  int s = (int)src[e];
  int d = (int)dst[e];
  float norm = dinv[s] * dinv[d];
  const float4 hv = *(const float4*)(h + (size_t)s * FDIM + c * 4);
  float* out = agg + (size_t)d * FDIM + c * 4;
  atomic_add_f32(out + 0, hv.x * norm);
  atomic_add_f32(out + 1, hv.y * norm);
  atomic_add_f32(out + 2, hv.z * norm);
  atomic_add_f32(out + 3, hv.w * norm);
}

// agg[i] += h[i] * dinv[i]^2 + bias   (self-loop term + bias)
__global__ __launch_bounds__(256) void selfloop_bias_kernel(
    float* __restrict__ agg, const float* __restrict__ h,
    const float* __restrict__ dinv, const float* __restrict__ bias, int n) {
  int gid = blockIdx.x * 256 + threadIdx.x;
  int node = gid >> 6;
  int c = gid & 63;
  if (node >= n) return;
  float di = dinv[node];
  float s = di * di;
  float4 a  = *(float4*)(agg + (size_t)node * FDIM + c * 4);
  float4 hv = *(const float4*)(h + (size_t)node * FDIM + c * 4);
  float4 b  = *(const float4*)(bias + c * 4);
  a.x += hv.x * s + b.x;
  a.y += hv.y * s + b.y;
  a.z += hv.z * s + b.z;
  a.w += hv.w * s + b.w;
  *(float4*)(agg + (size_t)node * FDIM + c * 4) = a;
}

// ---------------------------------------------------------------------------
// Pooling
// ---------------------------------------------------------------------------
__global__ void pool_init_kernel(unsigned* __restrict__ maxenc,
                                 float* __restrict__ sums,
                                 float* __restrict__ cnt) {
  int i = blockIdx.x * blockDim.x + threadIdx.x;
  if (i < NUM_GRAPHS_C * FDIM) {
    maxenc[i] = 0x007FFFFFu;  // fenc(-inf)
    sums[i] = 0.0f;
  }
  if (i < NUM_GRAPHS_C) cnt[i] = 0.0f;
}

__global__ __launch_bounds__(256) void pool_acc_kernel(
    const float* __restrict__ h, const long long* __restrict__ batch,
    unsigned* __restrict__ maxenc, float* __restrict__ sums,
    float* __restrict__ cnt, int n) {
  int node = blockIdx.x;
  if (node >= n) return;
  int g = (int)batch[node];
  int f = threadIdx.x;
  float v = h[(size_t)node * FDIM + f];
  __hip_atomic_fetch_max(&maxenc[g * FDIM + f], fenc(v), __ATOMIC_RELAXED,
                         __HIP_MEMORY_SCOPE_AGENT);
  atomic_add_f32(&sums[g * FDIM + f], v);
  if (f == 0) atomic_add_f32(&cnt[g], 1.0f);
}

__global__ void pool_final_kernel(const unsigned* __restrict__ maxenc,
                                  const float* __restrict__ sums,
                                  const float* __restrict__ cnt,
                                  float* __restrict__ out) {
  int g = blockIdx.x;
  int j = threadIdx.x;  // 0..511
  float r;
  if (j < FDIM) {
    r = fdec(maxenc[g * FDIM + j]);
  } else {
    float c = cnt[g];
    r = sums[g * FDIM + (j - FDIM)] / fmaxf(c, 1.0f);
  }
  out[(size_t)g * (2 * FDIM) + j] = r;
}

// ---------------------------------------------------------------------------
extern "C" void kernel_launch(void* const* d_in, const int* in_sizes, int n_in,
                              void* d_out, int out_size, void* d_ws,
                              size_t ws_size, hipStream_t stream) {
  (void)in_sizes; (void)n_in; (void)out_size; (void)ws_size;
  const float* x = (const float*)d_in[0];
  const long long* edge_index = (const long long*)d_in[1];
  const long long* batch = (const long long*)d_in[2];
  const float* W1 = (const float*)d_in[3];
  const float* b1 = (const float*)d_in[4];
  const float* W2 = (const float*)d_in[5];
  const float* b2 = (const float*)d_in[6];
  const float* W3 = (const float*)d_in[7];
  const float* b3 = (const float*)d_in[8];
  const long long* src = edge_index;             // edge_index[0]
  const long long* dst = edge_index + N_EDGES_C; // edge_index[1]
  float* out = (float*)d_out;

  // workspace carve-out (256B aligned regions)
  char* wsp = (char*)d_ws;
  auto carve = [&](size_t bytes) {
    char* p = wsp;
    wsp += (bytes + 255) & ~(size_t)255;
    return p;
  };
  float*    hbuf   = (float*)carve((size_t)N_NODES_C * FDIM * 4);
  float*    aggbuf = (float*)carve((size_t)N_NODES_C * FDIM * 4);
  float*    dinv   = (float*)carve((size_t)N_NODES_C * 4);
  int*      degi   = (int*)carve((size_t)N_NODES_C * 4);
  unsigned* maxenc = (unsigned*)carve((size_t)NUM_GRAPHS_C * FDIM * 4);
  float*    sums   = (float*)carve((size_t)NUM_GRAPHS_C * FDIM * 4);
  float*    cnt    = (float*)carve((size_t)NUM_GRAPHS_C * 4);

  // degree + d^-1/2
  hipMemsetAsync(degi, 0, (size_t)N_NODES_C * 4, stream);
  deg_kernel<<<(N_EDGES_C + 255) / 256, 256, 0, stream>>>(dst, degi, N_EDGES_C);
  dinv_kernel<<<(N_NODES_C + 255) / 256, 256, 0, stream>>>(degi, dinv, N_NODES_C);

  dim3 ggrid((N_NODES_C + 31) / 32, FDIM / 128);
  const int edge_blocks = (N_EDGES_C * 64) / 256;          // 200000
  const int node_blocks = (N_NODES_C * 64 + 255) / 256;    // 12500

  const float* in_feat = x;
  int K = IN_F_C;
  const float* Ws[3] = {W1, W2, W3};
  const float* bs[3] = {b1, b2, b3};
  for (int l = 0; l < 3; ++l) {
    gemm_wmma_bf16<<<ggrid, 256, 0, stream>>>(in_feat, Ws[l], hbuf, N_NODES_C, K);
    hipMemsetAsync(aggbuf, 0, (size_t)N_NODES_C * FDIM * 4, stream);
    edge_agg_kernel<<<edge_blocks, 256, 0, stream>>>(hbuf, src, dst, dinv,
                                                     aggbuf, N_EDGES_C);
    selfloop_bias_kernel<<<node_blocks, 256, 0, stream>>>(aggbuf, hbuf, dinv,
                                                          bs[l], N_NODES_C);
    in_feat = aggbuf;
    K = FDIM;
  }

  pool_init_kernel<<<NUM_GRAPHS_C, 256, 0, stream>>>(maxenc, sums, cnt);
  pool_acc_kernel<<<N_NODES_C, 256, 0, stream>>>(aggbuf, batch, maxenc, sums,
                                                 cnt, N_NODES_C);
  pool_final_kernel<<<NUM_GRAPHS_C, 512, 0, stream>>>(maxenc, sums, cnt, out);
}
